// AttentionEncoder_88287347736769
// MI455X (gfx1250) — compile-verified, matrix operand order checked by reading, not run on hardware
//
#include <hip/hip_runtime.h>
#include <hip/hip_bf16.h>

typedef __bf16 bf16_t;
typedef __bf16 v8bf  __attribute__((ext_vector_type(8)));
typedef __bf16 v16bf __attribute__((ext_vector_type(16)));
typedef float  v8f   __attribute__((ext_vector_type(8)));

#define DEV __device__ __forceinline__

constexpr int Lc = 4, Dc = 1024, Hc = 16, Bc = 4, Nc = 1024, DKc = 64;
constexpr float BN_EPS = 1e-5f;

DEV v8f wmma_bf16(v16bf a, v16bf b, v8f c) {
  return __builtin_amdgcn_wmma_f32_16x16x32_bf16(false, a, false, b, (short)0, c,
                                                 false, false);
}

// Fragment K-mapping (ISA 7.12.2): lane (half,r), element e holds
// K = (e<8 ? e : e+8) + 8*half ==> two contiguous 8-element runs; with K
// contiguous in the LDS row each half is one aligned ds_load_b128.
DEV v16bf load_frag(const bf16_t* rowp, int half) {
  v8bf lo = *(const v8bf*)(rowp + 8 * half);
  v8bf hi = *(const v8bf*)(rowp + 16 + 8 * half);
  return __builtin_shufflevector(lo, hi, 0, 1, 2, 3, 4, 5, 6, 7, 8, 9, 10, 11,
                                 12, 13, 14, 15);
}

// 16-lane butterfly reductions in the VALU via DPP16 (no LDS, co-executes
// with WMMA). The DPP control must be an immediate -> template parameter.
template <int CTRL>
DEV float dpp_bcast(float x) {
  int v = __builtin_amdgcn_update_dpp(0, __builtin_bit_cast(int, x), CTRL, 0xF,
                                      0xF, true);
  return __builtin_bit_cast(float, v);
}
DEV float red16_max(float x) {
  x = fmaxf(x, dpp_bcast<0xB1>(x));   // quad_perm(1,0,3,2)  : xor 1
  x = fmaxf(x, dpp_bcast<0x4E>(x));   // quad_perm(2,3,0,1)  : xor 2
  x = fmaxf(x, dpp_bcast<0x141>(x));  // row_half_mirror     : cross-quad
  x = fmaxf(x, dpp_bcast<0x140>(x));  // row_mirror          : cross-8
  return x;
}
DEV float red16_sum(float x) {
  x += dpp_bcast<0xB1>(x);
  x += dpp_bcast<0x4E>(x);
  x += dpp_bcast<0x141>(x);
  x += dpp_bcast<0x140>(x);
  return x;
}

// ---------------------------------------------------------------------------
__global__ void cvt_f32_bf16(const float* __restrict__ in, bf16_t* __restrict__ out,
                             int n) {
  int i = blockIdx.x * 256 + threadIdx.x;
  if (i < n) out[i] = (bf16_t)in[i];
}

__global__ void prep_x(const float* __restrict__ x, bf16_t* __restrict__ xb,
                       bf16_t* __restrict__ h1) {
  long long i = (long long)blockIdx.x * 256 + threadIdx.x;
  const long long total = (long long)Bc * Dc * Nc;
  if (i >= total) return;
  long long b = i / ((long long)Dc * Nc);
  long long rem = i - b * (long long)Dc * Nc;  // c*N + n
  bf16_t v = (bf16_t)x[i];
  xb[i] = v;
  h1[b * (2LL * Dc * Nc) + (long long)Dc * Nc + rem] = v;
}

// ---------------------------------------------------------------------------
// Tiled bf16 GEMM: C[M,N] = A[M,K]*B[K,N] (+epilogue). Block tile 128x128,
// K-stage 32, register-pipelined double-buffered LDS (1 barrier/stage, next
// stage's global loads in flight during WMMA). 8 waves (4x2); wave tile 32x64
// = 8 v_wmma per stage. A staged [m][k]; B staged transposed [n][k] so all
// fragment reads are 2x ds_load_b128.
// EPI: 0 bias->bf16 ; 1 bias+BN+ReLU->bf16 ; 2 bias+residual-add->f32 ;
//      3 (bias+acc)*0.125->bf16 (q projection, folds 1/sqrt(DK))
// ---------------------------------------------------------------------------
template <int EPI>
__global__ __launch_bounds__(256) void gemm_bf16_kernel(
    const bf16_t* __restrict__ A, const bf16_t* __restrict__ Bm,
    const float* __restrict__ bias, const float* __restrict__ bn_g,
    const float* __restrict__ bn_b, const float* __restrict__ bn_mu,
    const float* __restrict__ bn_var, bf16_t* __restrict__ outb,
    float* __restrict__ outf, int M, int N, int K, long long strideB,
    long long strideO, int rowOff) {
  __shared__ __align__(16) bf16_t As[2][128][40];  // [m][k]
  __shared__ __align__(16) bf16_t Bt[2][128][40];  // [n][k] transposed

  const int tid = threadIdx.x;
  const int wid = tid >> 5, lane = tid & 31;
  const int half = lane >> 4, r = lane & 15;
  const int wm = wid >> 1, wn = wid & 1;  // 4x2 wave grid
  const int m0 = blockIdx.y * 128, n0 = blockIdx.x * 128;
  const int z = blockIdx.z;
  const bf16_t* Bp = Bm + (long long)z * strideB;

  v8f acc[2][4];
#pragma unroll
  for (int i = 0; i < 2; ++i)
#pragma unroll
    for (int j = 0; j < 4; ++j) acc[i][j] = {};

  const int a_c8 = (tid & 3) * 8;   // 4 threads cover 32 K-cols
  const int a_r = tid >> 2;         // 64 rows/pass, 2 passes
  const int b_n8 = (tid & 15) * 8;  // 16 threads cover 128 n-cols
  const int b_k = tid >> 4;         // 16 k-rows/pass, 2 passes

  const bf16_t* aSrc = A + (long long)(m0 + a_r) * K + a_c8;
  const bf16_t* bSrc = Bp + (long long)b_k * N + n0 + b_n8;

  int4 ra[2];
  v8bf rb[2];
  // prologue: stage 0
#pragma unroll
  for (int p = 0; p < 2; ++p) {
    ra[p] = *(const int4*)(aSrc + (long long)(p * 64) * K);
    rb[p] = *(const v8bf*)(bSrc + (long long)(p * 16) * N);
  }
#pragma unroll
  for (int p = 0; p < 2; ++p) {
    *(int4*)&As[0][a_r + p * 64][a_c8] = ra[p];
#pragma unroll
    for (int j = 0; j < 8; ++j) Bt[0][b_n8 + j][b_k + p * 16] = rb[p][j];
  }
  __syncthreads();

  int cur = 0;
  for (int k0 = 0; k0 < K; k0 += 32, cur ^= 1) {
    const bool more = (k0 + 32 < K);
    if (more) {  // issue next stage's global loads before computing
#pragma unroll
      for (int p = 0; p < 2; ++p) {
        ra[p] = *(const int4*)(aSrc + (long long)(p * 64) * K + k0 + 32);
        rb[p] = *(const v8bf*)(bSrc + (long long)(k0 + 32 + p * 16) * N);
      }
      if (k0 + 64 < K) {  // global_prefetch_b8 two stages ahead
        __builtin_prefetch(aSrc + (long long)k0 + 64, 0, 0);
        __builtin_prefetch(bSrc + (long long)(k0 + 64) * N, 0, 0);
      }
    }

    v16bf af[2], bf[4];
#pragma unroll
    for (int sm = 0; sm < 2; ++sm)
      af[sm] = load_frag(&As[cur][wm * 32 + sm * 16 + r][0], half);
#pragma unroll
    for (int sn = 0; sn < 4; ++sn)
      bf[sn] = load_frag(&Bt[cur][wn * 64 + sn * 16 + r][0], half);
#pragma unroll
    for (int sm = 0; sm < 2; ++sm)
#pragma unroll
      for (int sn = 0; sn < 4; ++sn)
        acc[sm][sn] = wmma_bf16(af[sm], bf[sn], acc[sm][sn]);

    if (more) {  // store next stage into the alternate buffer
#pragma unroll
      for (int p = 0; p < 2; ++p) {
        *(int4*)&As[cur ^ 1][a_r + p * 64][a_c8] = ra[p];
#pragma unroll
        for (int j = 0; j < 8; ++j)
          Bt[cur ^ 1][b_n8 + j][b_k + p * 16] = rb[p][j];
      }
    }
    __syncthreads();
  }

  // Epilogue. C layout: VGPR i of lane (half,r) holds (M = i+8*half, N = r).
#pragma unroll
  for (int sm = 0; sm < 2; ++sm) {
#pragma unroll
    for (int i = 0; i < 8; ++i) {
      int m = m0 + wm * 32 + sm * 16 + half * 8 + i;
      float bs = bias[m];
      float g = 0.f, bb = 0.f, mu = 0.f, iv = 0.f;
      if (EPI == 1) {
        g = bn_g[m];
        bb = bn_b[m];
        mu = bn_mu[m];
        iv = rsqrtf(bn_var[m] + BN_EPS);
      }
#pragma unroll
      for (int sn = 0; sn < 4; ++sn) {
        int n = n0 + wn * 64 + sn * 16 + r;
        float vv = acc[sm][sn][i] + bs;
        if (EPI == 1) {
          vv = g * (vv - mu) * iv + bb;
          vv = vv > 0.f ? vv : 0.f;
        }
        if (EPI == 3) vv *= 0.125f;  // fold 1/sqrt(DK) into q
        if (EPI == 2) {
          float* p = outf + (long long)z * strideO + (long long)m * N + n;
          *p = *p + vv;
        } else {
          outb[(long long)z * strideO + (long long)(rowOff + m) * N + n] =
              (bf16_t)vv;
        }
      }
    }
  }
}

// ---------------------------------------------------------------------------
// Flash attention. q/k/v bf16 (B, D, N), channel = dk*H + h; q pre-scaled by
// 1/sqrt(DK). Grid (N/128, H, B); 8 waves, wave owns 16 queries. Per 32-key
// tile: 4 WMMA for S, online softmax via DPP16 butterflies, 4 WMMA for O.
// K/V double-buffered + register-pipelined (1 barrier per tile).
// ---------------------------------------------------------------------------
__global__ __launch_bounds__(256) void attn_kernel(const bf16_t* __restrict__ q,
                                                   const bf16_t* __restrict__ k,
                                                   const bf16_t* __restrict__ v,
                                                   bf16_t* __restrict__ attn) {
  __shared__ __align__(16) bf16_t Qs[128][72];       // [qrow][dk]
  __shared__ __align__(16) bf16_t Kt[2][32][72];     // [key][dk] (transposed)
  __shared__ __align__(16) bf16_t Vs[2][64][40];     // [dk][key]
  __shared__ __align__(16) bf16_t Ps[8][16][40];     // per-wave P patch

  const int tid = threadIdx.x;
  const int wid = tid >> 5, lane = tid & 31;
  const int half = lane >> 4, r = lane & 15;
  const int h = blockIdx.y, b = blockIdx.z;
  const int qn0 = blockIdx.x * 128;
  const long long base = (long long)b * Dc * Nc;

  // stage Q tile: read 8 contiguous n per thread, scatter-transpose into Qs
  {
    const int dk = tid >> 2;        // 0..63
    const int nl8 = (tid & 3) * 8;  // 4 threads * 8 = 32 n per pass
#pragma unroll
    for (int p = 0; p < 4; ++p) {
      int nl = nl8 + p * 32;
      v8bf t = *(const v8bf*)(q + base + (long long)(dk * Hc + h) * Nc + qn0 + nl);
#pragma unroll
      for (int j = 0; j < 8; ++j) Qs[nl + j][dk] = t[j];
    }
  }

  const int kv_dk = tid >> 2;       // 0..63
  const int kv_m8 = (tid & 3) * 8;  // 0,8,16,24
  const long long kvoff = base + (long long)(kv_dk * Hc + h) * Nc + kv_m8;

  // prologue: stage key-tile 0
  v8bf rk = *(const v8bf*)(k + kvoff);
  v8bf rv = *(const v8bf*)(v + kvoff);
#pragma unroll
  for (int j = 0; j < 8; ++j) Kt[0][kv_m8 + j][kv_dk] = rk[j];
  *(v8bf*)&Vs[0][kv_dk][kv_m8] = rv;
  __syncthreads();

  v16bf aQ[2];
#pragma unroll
  for (int ks = 0; ks < 2; ++ks)
    aQ[ks] = load_frag(&Qs[wid * 16 + r][ks * 32], half);

  float mrow[8], lrow[8];
  v8f O[4];
#pragma unroll
  for (int i = 0; i < 8; ++i) { mrow[i] = -1e30f; lrow[i] = 0.f; }
#pragma unroll
  for (int t = 0; t < 4; ++t) O[t] = {};

  int cur = 0;
  for (int m0 = 0; m0 < Nc; m0 += 32, cur ^= 1) {
    const bool more = (m0 + 32 < Nc);
    if (more) {
      rk = *(const v8bf*)(k + kvoff + m0 + 32);
      rv = *(const v8bf*)(v + kvoff + m0 + 32);
      if (m0 + 64 < Nc) {
        __builtin_prefetch(k + kvoff + m0 + 64, 0, 0);
        __builtin_prefetch(v + kvoff + m0 + 64, 0, 0);
      }
    }

    // S = Qscaled^T K : two 16x16 key sub-tiles, K=64 over 2 k-steps
    v8f S0 = {}, S1 = {};
#pragma unroll
    for (int ks = 0; ks < 2; ++ks)
      S0 = wmma_bf16(aQ[ks], load_frag(&Kt[cur][r][ks * 32], half), S0);
#pragma unroll
    for (int ks = 0; ks < 2; ++ks)
      S1 = wmma_bf16(aQ[ks], load_frag(&Kt[cur][16 + r][ks * 32], half), S1);

    // online softmax; row (8*half+i) lives entirely in this 16-lane half
#pragma unroll
    for (int i = 0; i < 8; ++i) {
      float s0 = S0[i], s1 = S1[i];
      float mx = red16_max(fmaxf(s0, s1));
      float mnew = fmaxf(mrow[i], mx);
      float sc = __expf(mrow[i] - mnew);
      float p0 = __expf(s0 - mnew), p1 = __expf(s1 - mnew);
      float rs = red16_sum(p0 + p1);
      lrow[i] = lrow[i] * sc + rs;
      mrow[i] = mnew;
#pragma unroll
      for (int t = 0; t < 4; ++t) O[t][i] *= sc;
      Ps[wid][half * 8 + i][r] = (bf16_t)p0;
      Ps[wid][half * 8 + i][16 + r] = (bf16_t)p1;
    }

    // C-layout -> A-layout re-swizzle via per-wave LDS (same-wave DS in order)
    v16bf aP = load_frag(&Ps[wid][r][0], half);
#pragma unroll
    for (int t = 0; t < 4; ++t) {
      v16bf bV = load_frag(&Vs[cur][t * 16 + r][0], half);  // (K=m, col=dk)
      O[t] = wmma_bf16(aP, bV, O[t]);
    }

    if (more) {  // store next key-tile into the alternate buffer
#pragma unroll
      for (int j = 0; j < 8; ++j) Kt[cur ^ 1][kv_m8 + j][kv_dk] = rk[j];
      *(v8bf*)&Vs[cur ^ 1][kv_dk][kv_m8] = rv;
    }
    __syncthreads();
  }

  // normalize and write attn (channel = dk*H + h)
#pragma unroll
  for (int t = 0; t < 4; ++t) {
#pragma unroll
    for (int i = 0; i < 8; ++i) {
      float o = O[t][i] / lrow[i];
      int qidx = qn0 + wid * 16 + half * 8 + i;
      int dk = t * 16 + r;
      attn[base + (long long)(dk * Hc + h) * Nc + qidx] = (bf16_t)o;
    }
  }
}

// ---------------------------------------------------------------------------
extern "C" void kernel_launch(void* const* d_in, const int* in_sizes, int n_in,
                              void* d_out, int out_size, void* d_ws,
                              size_t ws_size, hipStream_t stream) {
  (void)in_sizes; (void)n_in; (void)out_size; (void)ws_size;
  const float* x0     = (const float*)d_in[0];
  const float* Wq     = (const float*)d_in[1];
  const float* bq     = (const float*)d_in[2];
  const float* Wk     = (const float*)d_in[3];
  const float* bk     = (const float*)d_in[4];
  const float* Wv     = (const float*)d_in[5];
  const float* bv     = (const float*)d_in[6];
  const float* Wm     = (const float*)d_in[7];
  const float* bm     = (const float*)d_in[8];
  const float* Wp1    = (const float*)d_in[9];
  const float* bp1    = (const float*)d_in[10];
  const float* bn_g   = (const float*)d_in[11];
  const float* bn_b   = (const float*)d_in[12];
  const float* bn_mu  = (const float*)d_in[13];
  const float* bn_var = (const float*)d_in[14];
  const float* Wp2    = (const float*)d_in[15];
  const float* bp2    = (const float*)d_in[16];
  float* out = (float*)d_out;

  char* ws = (char*)d_ws;
  size_t off = 0;
  auto take = [&](size_t bytes) -> char* {
    char* p = ws + off;
    off = (off + bytes + 255) & ~(size_t)255;
    return p;
  };
  const size_t act = (size_t)Bc * Dc * Nc;  // 4M elements
  bf16_t* xb    = (bf16_t*)take(act * 2);
  bf16_t* qb    = (bf16_t*)take(act * 2);
  bf16_t* kb    = (bf16_t*)take(act * 2);
  bf16_t* vb    = (bf16_t*)take(act * 2);
  bf16_t* attnb = (bf16_t*)take(act * 2);
  bf16_t* h1    = (bf16_t*)take(act * 4);  // (B, 2D, N)
  bf16_t* h2    = (bf16_t*)take(act * 4);
  const int DD = Dc * Dc;
  bf16_t* wqb  = (bf16_t*)take((size_t)DD * 2);
  bf16_t* wkb  = (bf16_t*)take((size_t)DD * 2);
  bf16_t* wvb  = (bf16_t*)take((size_t)DD * 2);
  bf16_t* wmb  = (bf16_t*)take((size_t)DD * 2);
  bf16_t* wp1b = (bf16_t*)take((size_t)4 * DD * 2);
  bf16_t* wp2b = (bf16_t*)take((size_t)2 * DD * 2);

  (void)hipMemcpyAsync(out, x0, act * sizeof(float), hipMemcpyDeviceToDevice,
                       stream);

  const dim3 g88(Nc / 128, Dc / 128, Bc);
  const dim3 g8x16(Nc / 128, (2 * Dc) / 128, Bc);
  const dim3 gAttn(Nc / 128, Hc, Bc);
  const long long sD = (long long)Dc * Nc, s2D = 2LL * Dc * Nc;

  for (int l = 0; l < Lc; ++l) {
    cvt_f32_bf16<<<(DD + 255) / 256, 256, 0, stream>>>(Wq + (size_t)l * DD, wqb, DD);
    cvt_f32_bf16<<<(DD + 255) / 256, 256, 0, stream>>>(Wk + (size_t)l * DD, wkb, DD);
    cvt_f32_bf16<<<(DD + 255) / 256, 256, 0, stream>>>(Wv + (size_t)l * DD, wvb, DD);
    cvt_f32_bf16<<<(DD + 255) / 256, 256, 0, stream>>>(Wm + (size_t)l * DD, wmb, DD);
    cvt_f32_bf16<<<(4 * DD + 255) / 256, 256, 0, stream>>>(Wp1 + (size_t)l * 4 * DD,
                                                           wp1b, 4 * DD);
    cvt_f32_bf16<<<(2 * DD + 255) / 256, 256, 0, stream>>>(Wp2 + (size_t)l * 2 * DD,
                                                           wp2b, 2 * DD);
    prep_x<<<(unsigned)((act + 255) / 256), 256, 0, stream>>>(out, xb, h1);

    // q projection folds the 1/sqrt(DK) scale (EPI=3)
    gemm_bf16_kernel<3><<<g88, 256, 0, stream>>>(
        wqb, xb, bq + l * Dc, nullptr, nullptr, nullptr, nullptr, qb, nullptr,
        Dc, Nc, Dc, sD, sD, 0);
    gemm_bf16_kernel<0><<<g88, 256, 0, stream>>>(
        wkb, xb, bk + l * Dc, nullptr, nullptr, nullptr, nullptr, kb, nullptr,
        Dc, Nc, Dc, sD, sD, 0);
    gemm_bf16_kernel<0><<<g88, 256, 0, stream>>>(
        wvb, xb, bv + l * Dc, nullptr, nullptr, nullptr, nullptr, vb, nullptr,
        Dc, Nc, Dc, sD, sD, 0);

    attn_kernel<<<gAttn, 256, 0, stream>>>(qb, kb, vb, attnb);

    gemm_bf16_kernel<0><<<g88, 256, 0, stream>>>(
        wmb, attnb, bm + l * Dc, nullptr, nullptr, nullptr, nullptr, h1,
        nullptr, Dc, Nc, Dc, sD, s2D, 0);

    gemm_bf16_kernel<1><<<g8x16, 256, 0, stream>>>(
        wp1b, h1, bp1 + l * 2 * Dc, bn_g + l * 2 * Dc, bn_b + l * 2 * Dc,
        bn_mu + l * 2 * Dc, bn_var + l * 2 * Dc, h2, nullptr, 2 * Dc, Nc,
        2 * Dc, s2D, s2D, 0);

    gemm_bf16_kernel<2><<<g88, 256, 0, stream>>>(
        wp2b, h2, bp2 + l * Dc, nullptr, nullptr, nullptr, nullptr, nullptr,
        out, Dc, Nc, 2 * Dc, s2D, sD, 0);
  }
}
